// MPN_30520037605937
// MI455X (gfx1250) — compile-verified, matrix operand order checked by reading, not run on hardware
//
#include <hip/hip_runtime.h>
#include <hip/hip_bf16.h>

// ---------------------------------------------------------------------------
// MPN forward for MI455X (gfx1250, wave32, WMMA).
// All linears -> bf16 WMMA GEMM (v_wmma_f32_16x16x32_bf16), f32 accumulate.
// Gathers fused as row-maps on A; concats as segmented-K with all segment
// boundaries padded to multiples of BK=32 so staging is pure b128 traffic.
// Weights pre-transposed to [N,Kp] for contiguous B-tile staging.
// ---------------------------------------------------------------------------

typedef __bf16 bf16_t;
typedef __attribute__((ext_vector_type(16))) __bf16 v16bf;
typedef __attribute__((ext_vector_type(8)))  float  v8f;
typedef unsigned int u32x4 __attribute__((ext_vector_type(4)));

#define HIDN     256
#define NHEAD    4
#define MAXNEI   10
#define BATCH    128
#define NATOM    128
#define BN_TOT   (BATCH * NATOM)          // 16384
#define RROWS    (BN_TOT * MAXNEI)        // 163840

// ---------------- GEMM: C[M,N] = act(A[M,Ktot] @ W[Ktot,N] + b) -------------
// Requirements (guaranteed by launcher): M % 128 == 0, N % 64 == 0,
// s0.K % 32 == 0, Ktot % 32 == 0, all row bases 16B aligned.
struct Seg {
    const bf16_t* A;   // bf16 source rows
    const int*    map; // optional row remap (gather), else identity
    int           K;   // columns of this segment (multiple of 32)
    int           ld;  // leading dim of A (multiple of 32)
};

#define BM 128
#define BN 64
#define BK 32

__launch_bounds__(256)
__global__ void wmma_gemm_k(Seg s0, Seg s1,
                            const bf16_t* __restrict__ Wt, int Ktot, int N,
                            const float*  __restrict__ bias,
                            const bf16_t* __restrict__ emul,     // optional [M,N] elementwise mul
                            const float*  __restrict__ rowscale, // optional [M] scale
                            float* outF, bf16_t* outB,
                            int M, int relu)
{
    __shared__ __align__(16) bf16_t As[BM * BK];   // [row][k]
    __shared__ __align__(16) bf16_t Bs[BN * BK];   // [col][k] (K-major)

    const int tid  = threadIdx.x;
    const int lane = tid & 31;
    const int wid  = tid >> 5;
    const int wm0  = (wid & 3) * 32;
    const int wn0  = (wid >> 2) * 32;
    const int rowBase = blockIdx.x * BM;
    const int colBase = blockIdx.y * BN;

    // A staging: each thread covers 16 contiguous halves (32B) of one row.
    const int arow  = tid >> 1;          // 0..127
    const int akOff = (tid & 1) * 16;
    const int gRow  = rowBase + arow;
    const bf16_t* aptr0 = s0.A + (long)(s0.map ? s0.map[gRow] : gRow) * s0.ld;
    const bf16_t* aptr1 = s1.A ? (s1.A + (long)(s1.map ? s1.map[gRow] : gRow) * s1.ld)
                               : (const bf16_t*)nullptr;
    // B staging: each thread covers 8 contiguous halves (16B) of one W column.
    const int bcol  = tid >> 2;          // 0..63
    const int bkOff = (tid & 3) * 8;
    const bf16_t* wrow = Wt + (long)(colBase + bcol) * Ktot;

    v8f c00 = {}, c01 = {}, c10 = {}, c11 = {};

    for (int k0 = 0; k0 < Ktot; k0 += BK) {
        // ---- stage A tile: segment selection is uniform per k-step ----
        const bf16_t* src = (k0 < s0.K) ? (aptr0 + k0) : (aptr1 + (k0 - s0.K));
        {
            const u32x4* sv = (const u32x4*)(src + akOff);
            u32x4 va0 = sv[0], va1 = sv[1];
            u32x4* dv = (u32x4*)&As[arow * BK + akOff];
            dv[0] = va0; dv[1] = va1;
        }
        // ---- stage W tile (pre-transposed, contiguous) ----
        {
            u32x4 wv = *(const u32x4*)(wrow + k0 + bkOff);
            *(u32x4*)&Bs[bcol * BK + bkOff] = wv;
        }
        if (k0 + BK < Ktot) {
            const bf16_t* nsrc = (k0 + BK < s0.K) ? (aptr0 + k0 + BK) : (aptr1 + (k0 + BK - s0.K));
            __builtin_prefetch(nsrc + akOff, 0, 3);     // global_prefetch_b8
        }
        __syncthreads();

        // ---- build fragments per CDNA5 16-bit A/B layout ----
        // lane<16: K {0..7,16..23}; lane>=16: K {8..15,24..31}
        const int kb = (lane < 16) ? 0 : 8;
        union Frag { struct { u32x4 lo, hi; } q; v16bf v; };
        Frag a0, a1, b0, b1;
        {
            int r = wm0 + (lane & 15);
            const u32x4* p = (const u32x4*)&As[r * BK + kb];
            a0.q.lo = p[0]; a0.q.hi = p[2];
            const u32x4* p2 = (const u32x4*)&As[(r + 16) * BK + kb];
            a1.q.lo = p2[0]; a1.q.hi = p2[2];
            int c = wn0 + (lane & 15);
            const u32x4* q = (const u32x4*)&Bs[c * BK + kb];
            b0.q.lo = q[0]; b0.q.hi = q[2];
            const u32x4* q2 = (const u32x4*)&Bs[(c + 16) * BK + kb];
            b1.q.lo = q2[0]; b1.q.hi = q2[2];
        }
        c00 = __builtin_amdgcn_wmma_f32_16x16x32_bf16(false, a0.v, false, b0.v, (short)0, c00, false, false);
        c01 = __builtin_amdgcn_wmma_f32_16x16x32_bf16(false, a0.v, false, b1.v, (short)0, c01, false, false);
        c10 = __builtin_amdgcn_wmma_f32_16x16x32_bf16(false, a1.v, false, b0.v, (short)0, c10, false, false);
        c11 = __builtin_amdgcn_wmma_f32_16x16x32_bf16(false, a1.v, false, b1.v, (short)0, c11, false, false);
        __syncthreads();
    }

    // ---- epilogue: bias / emul / rowscale / relu, f32 or bf16 out ----
    auto store_tile = [&](v8f cc, int mOff, int nOff) {
        int col = colBase + wn0 + nOff + (lane & 15);
        int rbase = rowBase + wm0 + mOff + ((lane >= 16) ? 8 : 0);
        float bcol = bias[col];
        #pragma unroll
        for (int v = 0; v < 8; ++v) {
            int row = rbase + v;
            float f = cc[v] + bcol;
            long o = (long)row * N + col;
            if (emul)     f *= (float)emul[o];
            if (rowscale) f *= rowscale[row];
            if (relu)     f = f > 0.f ? f : 0.f;
            if (outF) outF[o] = f; else outB[o] = (bf16_t)f;
        }
    };
    store_tile(c00, 0, 0);   store_tile(c01, 0, 16);
    store_tile(c10, 16, 0);  store_tile(c11, 16, 16);
    (void)M;
}

// --------------------------- helper kernels --------------------------------
__global__ void f32_to_bf16_k(const float* __restrict__ in, bf16_t* __restrict__ out, long n) {
    long i = (long)blockIdx.x * blockDim.x + threadIdx.x;
    if (i < n) out[i] = (bf16_t)in[i];
}

// pad rows: in [M,Kin] f32 -> out [M,Kout] bf16 (cols >= Kin zeroed)
__global__ void pad_bf16_k(const float* __restrict__ in, bf16_t* __restrict__ out,
                           long M, int Kin, int Kout) {
    long t = (long)blockIdx.x * blockDim.x + threadIdx.x;
    if (t >= M * Kout) return;
    int  k = (int)(t % Kout);
    long r = t / Kout;
    out[t] = (k < Kin) ? (bf16_t)in[r * Kin + k] : (bf16_t)0.0f;
}

// transpose + pad: W [K,N] f32 -> out [N,Kp] bf16 (k >= K zeroed)
__global__ void wt_bf16_k(const float* __restrict__ W, bf16_t* __restrict__ out,
                          int K, int N, int Kp) {
    long t = (long)blockIdx.x * blockDim.x + threadIdx.x;
    if (t >= (long)N * Kp) return;
    int k = (int)(t % Kp);
    int n = (int)(t / Kp);
    out[t] = (k < K) ? (bf16_t)W[(long)k * N + n] : (bf16_t)0.0f;
}

__global__ void zero_f32_k(float* p, long n) {
    long i = (long)blockIdx.x * blockDim.x + threadIdx.x;
    if (i < n) p[i] = 0.f;
}

__global__ void build_maps_k(const int* __restrict__ ag, const int* __restrict__ bg,
                             const int* __restrict__ nnb,
                             int* __restrict__ amap, int* __restrict__ bmap,
                             float* __restrict__ rowmask) {
    int t = blockIdx.x * blockDim.x + threadIdx.x;
    if (t >= RROWS) return;
    int j  = t % MAXNEI;
    int bn = t / MAXNEI;
    amap[t] = ag[t * 2] * NATOM + ag[t * 2 + 1];
    bmap[t] = bg[t * 2] * NATOM + bg[t * 2 + 1];
    rowmask[t] = (j < nnb[bn]) ? 1.f : 0.f;
}

// f_nei[bn,c] = sum_j h_nei[bn*10+j, c]
__global__ void reduce_nei_k(const bf16_t* __restrict__ hnei, float* __restrict__ fnei) {
    long t = (long)blockIdx.x * blockDim.x + threadIdx.x;
    if (t >= (long)BN_TOT * HIDN) return;
    int bn = (int)(t / HIDN), c = (int)(t % HIDN);
    float s = 0.f;
    #pragma unroll
    for (int j = 0; j < MAXNEI; ++j)
        s += (float)hnei[((long)bn * MAXNEI + j) * HIDN + c];
    fnei[t] = s;
}

// out += f_nei * f_self * node_mask / DEPTH
__global__ void accum_layer_k(const float* __restrict__ fnei, const float* __restrict__ fself,
                              const float* __restrict__ nm, float* __restrict__ out) {
    long t = (long)blockIdx.x * blockDim.x + threadIdx.x;
    if (t >= (long)BN_TOT * HIDN) return;
    int bn = (int)(t / HIDN);
    out[t] += fnei[t] * fself[t] * nm[bn] * (1.f / 3.f);
}

// --------------------------- attention combine ------------------------------
// One wave32 per (b,n): neighbor softmax per head, head-gate softmax, scatter.
__launch_bounds__(128)
__global__ void attention_k(const float*  __restrict__ qout,   // [BN_TOT,256] f32
                            const bf16_t* __restrict__ kbuf,   // [RROWS,256]
                            const bf16_t* __restrict__ vbuf,
                            const bf16_t* __restrict__ ambuf,
                            const bf16_t* __restrict__ nlab,   // nei_label [RROWS,256]
                            const bf16_t* __restrict__ hnb,    // h_nei_bond [RROWS,256]
                            const bf16_t* __restrict__ feat,   // atom_feat [BN_TOT,256]
                            const bf16_t* __restrict__ gW, const float* __restrict__ gb,
                            bf16_t* __restrict__ fatt)
{
    int gw   = blockIdx.x * 4 + (threadIdx.x >> 5);
    int lane = threadIdx.x & 31;
    if (gw >= BN_TOT) return;
    int b = gw / NATOM, n = gw % NATOM;
    const long qb = (long)b * (NATOM * HIDN);           // per-batch offset, [N,256]
    const long rb = (long)b * ((long)NATOM * MAXNEI * HIDN);
    const int  d0 = lane * 2;                           // two d_k elems per lane

    float attn0[NHEAD], attn1[NHEAD], logit[NHEAD];

    for (int h = 0; h < NHEAD; ++h) {
        long qo = qb + (long)h * 8192 + n * 64 + d0;    // (B,H,N,64) C-order flat
        float q0 = qout[qo], q1 = qout[qo + 1];
        long kbase = rb + (long)h * 81920 + n * 640;    // (B,H,N,10,64) flat
        float sc[MAXNEI];
        for (int j = 0; j < MAXNEI; ++j) {
            long o = kbase + j * 64 + d0;
            float p = q0 * (float)kbuf[o] + q1 * (float)kbuf[o + 1];
            for (int off = 16; off; off >>= 1) p += __shfl_xor(p, off, 32);
            sc[j] = p;
        }
        float m = sc[0];
        for (int j = 1; j < MAXNEI; ++j) m = fmaxf(m, sc[j]);
        float ssum = 0.f;
        for (int j = 0; j < MAXNEI; ++j) { sc[j] = __expf(sc[j] - m); ssum += sc[j]; }
        float inv = 1.f / ssum;
        float a0 = 0.f, a1 = 0.f;
        for (int j = 0; j < MAXNEI; ++j) {
            long o = kbase + j * 64 + d0;
            float p = sc[j] * inv;
            a0 += p * (float)vbuf[o];
            a1 += p * (float)vbuf[o + 1];
        }
        attn0[h] = a0; attn1[h] = a1;

        // gate logit: dot( [cur_atom | element_max | ave], gW ) + gb
        float part = 0.f;
        #pragma unroll
        for (int t = 0; t < 8; ++t) {
            int idx = lane * 8 + t;      // 0..255
            float gi;
            if (idx < 64) {
                gi = (float)feat[qb + (long)h * 8192 + n * 64 + idx];
            } else if (idx < 128) {
                int d = idx - 64;
                long o = kbase + d;
                float mx = (float)ambuf[o];
                for (int j = 1; j < MAXNEI; ++j) mx = fmaxf(mx, (float)ambuf[o + j * 64]);
                gi = mx;
            } else {
                int d = idx - 128;       // 0..127 of 2*d_k
                float s = 0.f;
                for (int j = 0; j < MAXNEI; ++j) {
                    long orig = (long)h * 163840 + n * 1280 + j * 128 + d; // (H,N,10,128) flat
                    int n2 = (int)(orig / 5120);
                    int rr = (int)(orig % 5120);
                    int j2 = rr / 512;
                    int c  = rr % 512;
                    long base = ((long)(b * NATOM + n2) * MAXNEI + j2) * HIDN;
                    s += (c < 256) ? (float)nlab[base + c] : (float)hnb[base + (c - 256)];
                }
                gi = s * 0.1f;           // mean over 10 neighbors
            }
            part += gi * (float)gW[idx];
        }
        for (int off = 16; off; off >>= 1) part += __shfl_xor(part, off, 32);
        logit[h] = part + gb[0];
    }

    // softmax over heads, scale, scatter to [B,N,256] layout
    float mm = fmaxf(fmaxf(logit[0], logit[1]), fmaxf(logit[2], logit[3]));
    float e[NHEAD]; float es = 0.f;
    for (int h = 0; h < NHEAD; ++h) { e[h] = __expf(logit[h] - mm); es += e[h]; }
    float ies = 1.f / es;
    for (int h = 0; h < NHEAD; ++h) {
        float g = e[h] * ies;
        long f = (long)h * 8192 + n * 64 + d0;
        fatt[qb + f]     = (bf16_t)(g * attn0[h]);
        fatt[qb + f + 1] = (bf16_t)(g * attn1[h]);
    }
}

// ------------------------------- launch -------------------------------------
extern "C" void kernel_launch(void* const* d_in, const int* in_sizes, int n_in,
                              void* d_out, int out_size, void* d_ws, size_t ws_size,
                              hipStream_t stream) {
    const float* input_atom = (const float*)d_in[0];
    const float* input_bond = (const float*)d_in[1];
    const float* node_mask  = (const float*)d_in[2];
    const int*   atom_graph = (const int*)d_in[3];
    const int*   bond_graph = (const int*)d_in[4];
    const int*   num_nbs    = (const int*)d_in[5];
    const float* atomW = (const float*)d_in[6],  *atomB = (const float*)d_in[7];
    const float* neiAW = (const float*)d_in[8],  *neiAB = (const float*)d_in[9];
    const float* neiBW = (const float*)d_in[10], *neiBB = (const float*)d_in[11];
    const float* selfW = (const float*)d_in[12], *selfB = (const float*)d_in[13];
    const float* U2W   = (const float*)d_in[14], *U2B   = (const float*)d_in[15];
    const float* U1W   = (const float*)d_in[16], *U1B   = (const float*)d_in[17];
    const float* qW    = (const float*)d_in[18], *qB    = (const float*)d_in[19];
    const float* kW    = (const float*)d_in[20], *kB    = (const float*)d_in[21];
    const float* vW    = (const float*)d_in[22], *vB    = (const float*)d_in[23];
    const float* amW   = (const float*)d_in[24], *amB   = (const float*)d_in[25];
    const float* gW    = (const float*)d_in[26], *gB    = (const float*)d_in[27];
    float* out = (float*)d_out;

    // ---- workspace carve ----
    char* ws = (char*)d_ws;
    auto alloc = [&](size_t bytes) { void* p = (void*)ws; ws += (bytes + 255) & ~(size_t)255; return p; };
    // weights, transposed [N,Kp] bf16
    bf16_t* wAtomT = (bf16_t*)alloc((size_t)256 * 64  * 2);
    bf16_t* wNeiAT = (bf16_t*)alloc((size_t)256 * 256 * 2);
    bf16_t* wNeiBT = (bf16_t*)alloc((size_t)256 * 64  * 2);
    bf16_t* wSelfT = (bf16_t*)alloc((size_t)256 * 256 * 2);
    bf16_t* wU2T   = (bf16_t*)alloc((size_t)256 * 320 * 2);
    bf16_t* wU1T   = (bf16_t*)alloc((size_t)256 * 512 * 2);
    bf16_t* wQT    = (bf16_t*)alloc((size_t)256 * 256 * 2);
    bf16_t* wKT    = (bf16_t*)alloc((size_t)256 * 512 * 2);
    bf16_t* wVT    = (bf16_t*)alloc((size_t)256 * 512 * 2);
    bf16_t* wAmT   = (bf16_t*)alloc((size_t)256 * 512 * 2);
    bf16_t* wGv    = (bf16_t*)alloc((size_t)256 * 2);
    // padded bf16 inputs
    bf16_t* xAtom = (bf16_t*)alloc((size_t)BN_TOT * 64 * 2);
    bf16_t* xBond = (bf16_t*)alloc((size_t)BN_TOT * 64 * 2);
    int*    amap  = (int*)alloc((size_t)RROWS * 4);
    int*    bmap  = (int*)alloc((size_t)RROWS * 4);
    float*  rmask = (float*)alloc((size_t)RROWS * 4);
    bf16_t* featA = (bf16_t*)alloc((size_t)BN_TOT * HIDN * 2);
    bf16_t* featB = (bf16_t*)alloc((size_t)BN_TOT * HIDN * 2);
    bf16_t* fatt  = (bf16_t*)alloc((size_t)BN_TOT * HIDN * 2);
    bf16_t* hnb   = (bf16_t*)alloc((size_t)RROWS * HIDN * 2);
    bf16_t* big0  = (bf16_t*)alloc((size_t)RROWS * HIDN * 2);  // nei_label
    bf16_t* big1  = (bf16_t*)alloc((size_t)RROWS * HIDN * 2);  // h_nei -> k-out
    bf16_t* big2  = (bf16_t*)alloc((size_t)RROWS * HIDN * 2);  // v-out
    bf16_t* big3  = (bf16_t*)alloc((size_t)RROWS * HIDN * 2);  // am-out
    float*  fnei  = (float*)alloc((size_t)BN_TOT * HIDN * 4);
    float*  fself = (float*)alloc((size_t)BN_TOT * HIDN * 4);
    float*  qout  = (float*)alloc((size_t)BN_TOT * HIDN * 4);

    auto wt = [&](const float* s, bf16_t* d, int K, int N, int Kp) {
        long n = (long)N * Kp;
        wt_bf16_k<<<dim3((unsigned)((n + 255) / 256)), 256, 0, stream>>>(s, d, K, N, Kp);
    };
    auto pad = [&](const float* s, bf16_t* d, long M, int Kin, int Kout) {
        long n = M * Kout;
        pad_bf16_k<<<dim3((unsigned)((n + 255) / 256)), 256, 0, stream>>>(s, d, M, Kin, Kout);
    };
    auto gemm = [&](Seg s0, Seg s1, const bf16_t* Wt, int Ktot, int N, const float* bias,
                    const bf16_t* emul, const float* rowscale,
                    float* oF, bf16_t* oB, int M, int relu) {
        dim3 grid((M + BM - 1) / BM, (N + BN - 1) / BN);
        wmma_gemm_k<<<grid, 256, 0, stream>>>(s0, s1, Wt, Ktot, N, bias, emul, rowscale, oF, oB, M, relu);
    };
    const Seg noseg{nullptr, nullptr, 0, 0};

    // ---- one-time prep (per launch; deterministic) ----
    zero_f32_k<<<dim3((BN_TOT * HIDN + 255) / 256), 256, 0, stream>>>(out, (long)BN_TOT * HIDN);
    wt(atomW, wAtomT, 34,  256, 64);
    wt(neiAW, wNeiAT, 256, 256, 256);
    wt(neiBW, wNeiBT, 46,  256, 64);
    wt(selfW, wSelfT, 256, 256, 256);
    wt(U2W,   wU2T,   302, 256, 320);
    wt(U1W,   wU1T,   512, 256, 512);
    wt(qW,    wQT,    256, 256, 256);
    wt(kW,    wKT,    512, 256, 512);
    wt(vW,    wVT,    512, 256, 512);
    wt(amW,   wAmT,   512, 256, 512);
    f32_to_bf16_k<<<dim3(1), 256, 0, stream>>>(gW, wGv, 256);
    pad(input_atom, xAtom, BN_TOT, 34, 64);
    pad(input_bond, xBond, BN_TOT, 46, 64);
    build_maps_k<<<dim3((RROWS + 255) / 256), 256, 0, stream>>>(
        atom_graph, bond_graph, num_nbs, amap, bmap, rmask);

    // atom_feat = input_atom @ atom_W + atom_b    (K padded 34 -> 64)
    gemm(Seg{xAtom, nullptr, 64, 64}, noseg, wAtomT, 64, HIDN, atomB,
         nullptr, nullptr, nullptr, featA, BN_TOT, 0);
    // h_nei_bond = gather(input_bond)[bmap] @ neiB_W + neiB_b  (46 -> 64)
    gemm(Seg{xBond, bmap, 64, 64}, noseg, wNeiBT, 64, HIDN, neiBB,
         nullptr, nullptr, nullptr, hnb, RROWS, 0);

    bf16_t* feat = featA;
    bf16_t* next = featB;
    const long EW = (long)BN_TOT * HIDN;
    dim3 ewg((unsigned)((EW + 255) / 256));

    for (int depth = 0; depth < 3; ++depth) {
        // h_nei = (feat[amap] @ neiA_W + b) * h_nei_bond * mask   -> big1
        gemm(Seg{feat, amap, 256, 256}, noseg, wNeiAT, 256, HIDN, neiAB,
             hnb, rmask, nullptr, big1, RROWS, 0);
        reduce_nei_k<<<ewg, 256, 0, stream>>>(big1, fnei);
        // f_self
        gemm(Seg{feat, nullptr, 256, 256}, noseg, wSelfT, 256, HIDN, selfB,
             nullptr, nullptr, fself, nullptr, BN_TOT, 0);
        accum_layer_k<<<ewg, 256, 0, stream>>>(fnei, fself, node_mask, out);
        // nei_label = relu(concat(feat[amap], bond[bmap]) @ U2_W + b) * mask -> big0
        // (A cols 302..319 are xBond zero-pad; W rows 302..319 zero-padded too)
        gemm(Seg{feat, amap, 256, 256}, Seg{xBond, bmap, 64, 64}, wU2T, 320, HIDN, U2B,
             nullptr, rmask, nullptr, big0, RROWS, 1);
        // k / v / am on mh = concat(nei_label, h_nei_bond)  (K = 512)
        gemm(Seg{big0, nullptr, 256, 256}, Seg{hnb, nullptr, 256, 256}, wKT, 512, HIDN, kB,
             nullptr, nullptr, nullptr, big1, RROWS, 0);
        gemm(Seg{big0, nullptr, 256, 256}, Seg{hnb, nullptr, 256, 256}, wVT, 512, HIDN, vB,
             nullptr, nullptr, nullptr, big2, RROWS, 0);
        gemm(Seg{big0, nullptr, 256, 256}, Seg{hnb, nullptr, 256, 256}, wAmT, 512, HIDN, amB,
             nullptr, nullptr, nullptr, big3, RROWS, 0);
        // q = feat @ q_W + q_b (f32)
        gemm(Seg{feat, nullptr, 256, 256}, noseg, wQT, 256, HIDN, qB,
             nullptr, nullptr, qout, nullptr, BN_TOT, 0);
        // attention combine -> fatt
        attention_k<<<dim3(BN_TOT / 4), 128, 0, stream>>>(
            qout, big1, big2, big3, big0, hnb, feat, wGv, gB, fatt);
        // atom_feat = relu(concat(feat, fatt) @ U1_W + b)
        gemm(Seg{feat, nullptr, 256, 256}, Seg{fatt, nullptr, 256, 256}, wU1T, 512, HIDN, U1B,
             nullptr, nullptr, nullptr, next, BN_TOT, 1);
        bf16_t* t = feat; feat = next; next = t;
    }
    (void)in_sizes; (void)n_in; (void)out_size; (void)ws_size;
}
